// BiasedSelfAttentionLayer_68461778698649
// MI455X (gfx1250) — matrix operand order repaired and checked by reading, not compiled, not run on hardware
//
#include <hip/hip_runtime.h>

typedef __attribute__((ext_vector_type(16))) _Float16 v16h;
typedef __attribute__((ext_vector_type(8)))  float    v8f;
typedef unsigned int uv4  __attribute__((ext_vector_type(4)));
typedef unsigned int u32x4 __attribute__((ext_vector_type(4)));
typedef int          i32x8 __attribute__((ext_vector_type(8)));
typedef int          i32x4 __attribute__((ext_vector_type(4)));

#define HDIM 1024
#define BB   4
#define NSEQ 1024
#define NHEAD 16
#define HS   64
#define MROWS (BB * NSEQ)   // 4096
#define SCALE_F 0.125f      // 1/sqrt(64)

#if defined(__gfx1250__) && __has_builtin(__builtin_amdgcn_tensor_load_to_lds)
#define USE_TDM 1
#else
#define USE_TDM 0
#endif

union FragH { v16h v; uv4 q[2]; };

// ---------------- conversion kernels ----------------

__global__ void f32_to_f16_kernel(const float* __restrict__ in,
                                  _Float16* __restrict__ out, int n) {
  int i = blockIdx.x * blockDim.x + threadIdx.x;
  if (i < n) out[i] = (_Float16)in[i];
}

// W [k][n] f32 -> Wt [n][k] f16  (coalesced read along n)
__global__ void w_transpose_f16_kernel(const float* __restrict__ W,
                                       _Float16* __restrict__ Wt) {
  int i = blockIdx.x * blockDim.x + threadIdx.x;   // over 1024*1024
  int k = i >> 10, n = i & 1023;
  Wt[(size_t)n * HDIM + k] = (_Float16)W[i];
}

#if USE_TDM
// byte offset of a shared-memory pointer within the workgroup LDS allocation
__device__ inline unsigned lds_off(void* p) {
  return (unsigned)(unsigned long long)(__attribute__((address_space(3))) char*)p;
}

// TDM: DMA a 2D f16 tile [128 rows x 32 cols] (row stride 1024 elems) from
// global into LDS, padding each 64B row with 16B -> LDS row stride 40 f16.
__device__ inline void tdm_load_tile_f16(unsigned lds_byte_off,
                                         const _Float16* gptr,
                                         unsigned tensor_rows) {
  unsigned long long ga = (unsigned long long)(size_t)gptr;
  u32x4 g0;
  g0[0] = 1u;                                    // count=1, user desc
  g0[1] = lds_byte_off;                          // lds_addr
  g0[2] = (unsigned)(ga & 0xffffffffu);          // global_addr lo
  g0[3] = (unsigned)((ga >> 32) & 0x01ffffffu) | (2u << 30);  // addr hi | type=2
  i32x8 g1;
  // data_size=1 (2B), pad_enable, pad_interval=3 (16 DWORDs), pad_amount=3 (4 DWORDs)
  g1[0] = (int)((1u << 16) | (1u << 20) | (3u << 22) | (3u << 25));
  g1[1] = (int)((HDIM & 0xffffu) << 16);                       // tensor_dim0 lo16
  g1[2] = (int)((HDIM >> 16) | ((tensor_rows & 0xffffu) << 16)); // dim0 hi | dim1 lo
  g1[3] = (int)((tensor_rows >> 16) | (32u << 16));            // dim1 hi | tile_dim0=32
  g1[4] = (int)(128u);                                         // tile_dim1=128, tile_dim2=0
  g1[5] = (int)(HDIM);                                         // tensor_dim0_stride lo32
  g1[6] = 0;                                                   // stride hi16 | dim1_stride lo
  g1[7] = 0;
  i32x4 z4 = {};
  i32x8 z8 = {};
  __builtin_amdgcn_tensor_load_to_lds(g0, g1, z4, z4, z8, 0);
}
#endif

// ---------------- WMMA GEMM: C[M,1024] = A[M,1024] @ Wt^T + bias ----------------
// A f16 row-major [4096,1024]; Wt f16 [n][k] (pre-transposed weights).
// Workgroup tile 128x128, 8 waves, wave tile 32x64, K-step 32, double-buffered LDS.
// mode 0: out f32 [4096,1024]; mode 1: out f16 head-split [(b*16+h), n, d], *scale.

__launch_bounds__(256)
__global__ void gemm_f16_kernel(const _Float16* __restrict__ A,
                                const _Float16* __restrict__ Wt,
                                const float*    __restrict__ bias,
                                float*          __restrict__ outF,
                                _Float16*       __restrict__ outH,
                                int mode, float scale) {
  __shared__ __align__(16) _Float16 Asl[2][128][40];  // 128 m-rows x 32 k (+8 pad)
  __shared__ __align__(16) _Float16 Bsl[2][128][40];  // 128 n-rows x 32 k (+8 pad)

  const int tid  = threadIdx.x;
  const int lane = tid & 31, wave = tid >> 5;
  const int wm = wave >> 1, wn = wave & 1;            // 4x2 wave grid
  const int half = lane >> 4, l15 = lane & 15;
  const int n0 = blockIdx.x * 128;
  const int m0 = blockIdx.y * 128;

  v8f c[2][4];
  #pragma unroll
  for (int ms = 0; ms < 2; ++ms)
    #pragma unroll
    for (int ns = 0; ns < 4; ++ns) c[ms][ns] = (v8f){};

#if USE_TDM
  unsigned ldsA0 = lds_off(&Asl[0][0][0]), ldsA1 = lds_off(&Asl[1][0][0]);
  unsigned ldsB0 = lds_off(&Bsl[0][0][0]), ldsB1 = lds_off(&Bsl[1][0][0]);
  if (wave == 0) {
    tdm_load_tile_f16(ldsA0, A  + (size_t)m0 * HDIM, MROWS);
    tdm_load_tile_f16(ldsB0, Wt + (size_t)n0 * HDIM, HDIM);
    __builtin_amdgcn_s_wait_tensorcnt(0);
  }
  __syncthreads();
#else
  const int arow = tid >> 1, ah = tid & 1;  // loaders: 128 rows x 2 chunks of 16 f16
  {
    const uv4* srcA = (const uv4*)(A + (size_t)(m0 + arow) * HDIM + ah * 16);
    uv4* dstA = (uv4*)&Asl[0][arow][ah * 16];
    dstA[0] = srcA[0]; dstA[1] = srcA[1];
    const uv4* srcB = (const uv4*)(Wt + (size_t)(n0 + arow) * HDIM + ah * 16);
    uv4* dstB = (uv4*)&Bsl[0][arow][ah * 16];
    dstB[0] = srcB[0]; dstB[1] = srcB[1];
  }
  __syncthreads();
#endif

  for (int kt = 0; kt < HDIM; kt += 32) {
    const int cur = (kt >> 5) & 1;
    const int nxt = cur ^ 1;
    // prefetch next K-tile into the other buffer while computing this one
    if (kt + 32 < HDIM) {
#if USE_TDM
      if (wave == 0) {
        tdm_load_tile_f16(nxt ? ldsA1 : ldsA0, A  + (size_t)m0 * HDIM + kt + 32, MROWS);
        tdm_load_tile_f16(nxt ? ldsB1 : ldsB0, Wt + (size_t)n0 * HDIM + kt + 32, HDIM);
      }
#else
      const uv4* srcA = (const uv4*)(A + (size_t)(m0 + arow) * HDIM + kt + 32 + ah * 16);
      uv4* dstA = (uv4*)&Asl[nxt][arow][ah * 16];
      dstA[0] = srcA[0]; dstA[1] = srcA[1];
      const uv4* srcB = (const uv4*)(Wt + (size_t)(n0 + arow) * HDIM + kt + 32 + ah * 16);
      uv4* dstB = (uv4*)&Bsl[nxt][arow][ah * 16];
      dstB[0] = srcB[0]; dstB[1] = srcB[1];
#endif
    }

    FragH a0, a1;
    {
      const _Float16* p = &Asl[cur][wm * 32 + l15][0];
      a0.q[0] = *(const uv4*)(p + half * 8);
      a0.q[1] = *(const uv4*)(p + 16 + half * 8);
      const _Float16* p2 = &Asl[cur][wm * 32 + 16 + l15][0];
      a1.q[0] = *(const uv4*)(p2 + half * 8);
      a1.q[1] = *(const uv4*)(p2 + 16 + half * 8);
    }
    #pragma unroll
    for (int ns = 0; ns < 4; ++ns) {
      FragH b;
      const _Float16* p = &Bsl[cur][wn * 64 + ns * 16 + l15][0];
      b.q[0] = *(const uv4*)(p + half * 8);
      b.q[1] = *(const uv4*)(p + 16 + half * 8);
      c[0][ns] = __builtin_amdgcn_wmma_f32_16x16x32_f16(false, a0.v, false, b.v, (short)0, c[0][ns], false, false);
      c[1][ns] = __builtin_amdgcn_wmma_f32_16x16x32_f16(false, a1.v, false, b.v, (short)0, c[1][ns], false, false);
    }

#if USE_TDM
    if (wave == 0) __builtin_amdgcn_s_wait_tensorcnt(0);
#endif
    __syncthreads();
  }

  // epilogue: C layout elem e -> local row = e + 8*half, col = l15
  #pragma unroll
  for (int ms = 0; ms < 2; ++ms) {
    #pragma unroll
    for (int ns = 0; ns < 4; ++ns) {
      v8f cv = c[ms][ns];
      int rowbase = m0 + wm * 32 + ms * 16 + 8 * half;
      int col = n0 + wn * 64 + ns * 16 + l15;
      float bv = bias[col];
      #pragma unroll
      for (int e = 0; e < 8; ++e) {
        int row = rowbase + e;
        float val = cv[e] + bv;
        if (mode == 0) {
          outF[(size_t)row * HDIM + col] = val;
        } else {
          val *= scale;
          int b = row >> 10, nseq = row & 1023;
          int h = col >> 6,  d = col & 63;
          outH[((size_t)(b * NHEAD + h) * NSEQ + nseq) * HS + d] = (_Float16)val;
        }
      }
    }
  }
}

// ---------------- flash attention ----------------
// Q,K,V f16 head-split [(b*16+h), n, d]; Q pre-scaled by SCALE.
// feats f16 [B*N, H] with col = h*64 + d.

__launch_bounds__(128)
__global__ void attn_kernel(const _Float16* __restrict__ Q,
                            const _Float16* __restrict__ K,
                            const _Float16* __restrict__ V,
                            const float*    __restrict__ bias,
                            _Float16*       __restrict__ feats) {
  __shared__ __align__(16) _Float16 Ksl[32][72];      // [k_local][d] (+pad)
  __shared__ __align__(16) _Float16 Vsl[64][40];      // transposed: [d][k_local] (+pad)
  __shared__ __align__(16) _Float16 Psl[4][16][32];   // per-wave P tile

  const int tid  = threadIdx.x;
  const int lane = tid & 31, wave = tid >> 5;
  const int half = lane >> 4, l15 = lane & 15;
  const int head  = blockIdx.x;                 // b*16 + h
  const int qbase = blockIdx.y * 64 + wave * 16;
  const int b = head >> 4, h = head & 15;

  const _Float16* Qp = Q + (size_t)head * NSEQ * HS;
  const _Float16* Kp = K + (size_t)head * NSEQ * HS;
  const _Float16* Vp = V + (size_t)head * NSEQ * HS;
  const float* biasp = bias + (size_t)head * NSEQ * NSEQ;

  // Q A-fragments held across the whole loop (d 0..31 and 32..63)
  FragH qa0, qa1;
  {
    const _Float16* p = Qp + (size_t)(qbase + l15) * HS;
    qa0.q[0] = *(const uv4*)(p + half * 16);
    qa0.q[1] = *(const uv4*)(p + half * 16 + 8);
    qa1.q[0] = *(const uv4*)(p + 32 + half * 16);
    qa1.q[1] = *(const uv4*)(p + 40 + half * 16);
  }

  float mrow[8], lrow[8];
  #pragma unroll
  for (int e = 0; e < 8; ++e) { mrow[e] = -1e30f; lrow[e] = 0.f; }
  v8f o0 = {}, o1 = {}, o2 = {}, o3 = {};

  const int kr = tid >> 2, kq = tid & 3;   // loaders: 32 rows x 4 chunks of 16

  for (int kb = 0; kb < NSEQ; kb += 32) {
    // cooperative: K block as-is, V block transposed into LDS
    {
      const uv4* src = (const uv4*)(Kp + (size_t)(kb + kr) * HS + kq * 16);
      uv4* dst = (uv4*)&Ksl[kr][kq * 16];
      dst[0] = src[0];
      dst[1] = src[1];
    }
    {
      const _Float16* src = Vp + (size_t)(kb + kr) * HS + kq * 16;
      #pragma unroll
      for (int j = 0; j < 16; ++j) Vsl[kq * 16 + j][kr] = src[j];
    }
    __syncthreads();

    // S tiles: keys [kb, kb+16) and [kb+16, kb+32)
    v8f s0 = {}, s1 = {};
    {
      FragH k0, k1;
      const _Float16* p = &Ksl[l15][half * 16];
      k0.q[0] = *(const uv4*)(p);      k0.q[1] = *(const uv4*)(p + 8);
      k1.q[0] = *(const uv4*)(p + 32); k1.q[1] = *(const uv4*)(p + 40);
      s0 = __builtin_amdgcn_wmma_f32_16x16x32_f16(false, qa0.v, false, k0.v, (short)0, s0, false, false);
      s0 = __builtin_amdgcn_wmma_f32_16x16x32_f16(false, qa1.v, false, k1.v, (short)0, s0, false, false);
    }
    {
      FragH k0, k1;
      const _Float16* p = &Ksl[16 + l15][half * 16];
      k0.q[0] = *(const uv4*)(p);      k0.q[1] = *(const uv4*)(p + 8);
      k1.q[0] = *(const uv4*)(p + 32); k1.q[1] = *(const uv4*)(p + 40);
      s1 = __builtin_amdgcn_wmma_f32_16x16x32_f16(false, qa0.v, false, k0.v, (short)0, s1, false, false);
      s1 = __builtin_amdgcn_wmma_f32_16x16x32_f16(false, qa1.v, false, k1.v, (short)0, s1, false, false);
    }

    // bias + online softmax (rows = 8*half + e, cols = l15 / 16+l15)
    const int qrow = qbase + 8 * half;
    const int kc0 = kb + l15, kc1 = kb + 16 + l15;
    float sv0[8], sv1[8], rmax[8], alpha[8], rsum[8];
    #pragma unroll
    for (int e = 0; e < 8; ++e) {
      sv0[e] = s0[e] + biasp[(size_t)(qrow + e) * NSEQ + kc0];
      sv1[e] = s1[e] + biasp[(size_t)(qrow + e) * NSEQ + kc1];
      rmax[e] = fmaxf(sv0[e], sv1[e]);
    }
    #pragma unroll
    for (int m = 1; m < 16; m <<= 1)
      #pragma unroll
      for (int e = 0; e < 8; ++e) rmax[e] = fmaxf(rmax[e], __shfl_xor(rmax[e], m, 32));
    #pragma unroll
    for (int e = 0; e < 8; ++e) {
      float mn = fmaxf(mrow[e], rmax[e]);
      alpha[e] = __expf(mrow[e] - mn);
      mrow[e] = mn;
      sv0[e] = __expf(sv0[e] - mn);
      sv1[e] = __expf(sv1[e] - mn);
      rsum[e] = sv0[e] + sv1[e];
    }
    #pragma unroll
    for (int m = 1; m < 16; m <<= 1)
      #pragma unroll
      for (int e = 0; e < 8; ++e) rsum[e] += __shfl_xor(rsum[e], m, 32);
    #pragma unroll
    for (int e = 0; e < 8; ++e) {
      lrow[e] = lrow[e] * alpha[e] + rsum[e];
      o0[e] *= alpha[e]; o1[e] *= alpha[e]; o2[e] *= alpha[e]; o3[e] *= alpha[e];
    }

    // re-layout P (C layout -> A layout) via per-wave LDS
    #pragma unroll
    for (int e = 0; e < 8; ++e) {
      Psl[wave][8 * half + e][l15]      = (_Float16)sv0[e];
      Psl[wave][8 * half + e][16 + l15] = (_Float16)sv1[e];
    }
    asm volatile("s_wait_dscnt 0x0" ::: "memory");
    FragH pa;
    {
      const _Float16* p = &Psl[wave][l15][0];
      pa.q[0] = *(const uv4*)(p + half * 16);
      pa.q[1] = *(const uv4*)(p + half * 16 + 8);
    }

    // O += P @ V  (4 d-tiles of 16)
    #pragma unroll
    for (int t = 0; t < 4; ++t) {
      FragH vb;
      const _Float16* p = &Vsl[t * 16 + l15][half * 16];
      vb.q[0] = *(const uv4*)(p);
      vb.q[1] = *(const uv4*)(p + 8);
      v8f acc = (t == 0) ? o0 : (t == 1) ? o1 : (t == 2) ? o2 : o3;
      acc = __builtin_amdgcn_wmma_f32_16x16x32_f16(false, pa.v, false, vb.v, (short)0, acc, false, false);
      if (t == 0) o0 = acc; else if (t == 1) o1 = acc; else if (t == 2) o2 = acc; else o3 = acc;
    }
    __syncthreads();
  }

  // normalize and store feats [B*N, H], col = h*64 + t*16 + l15
  const int rowb = qbase + 8 * half;
  #pragma unroll
  for (int e = 0; e < 8; ++e) {
    float inv = 1.0f / lrow[e];
    _Float16* dst = feats + ((size_t)(b * NSEQ + rowb + e)) * HDIM + h * HS;
    dst[0 * 16 + l15] = (_Float16)(o0[e] * inv);
    dst[1 * 16 + l15] = (_Float16)(o1[e] * inv);
    dst[2 * 16 + l15] = (_Float16)(o2[e] * inv);
    dst[3 * 16 + l15] = (_Float16)(o3[e] * inv);
  }
}

// ---------------- launcher ----------------

extern "C" void kernel_launch(void* const* d_in, const int* in_sizes, int n_in,
                              void* d_out, int out_size, void* d_ws, size_t ws_size,
                              hipStream_t stream) {
  (void)in_sizes; (void)n_in; (void)out_size; (void)ws_size;
  const float* x   = (const float*)d_in[0];
  const float* ab  = (const float*)d_in[1];
  const float* Wq  = (const float*)d_in[2];
  const float* bq  = (const float*)d_in[3];
  const float* Wk  = (const float*)d_in[4];
  const float* bk  = (const float*)d_in[5];
  const float* Wv  = (const float*)d_in[6];
  const float* bv  = (const float*)d_in[7];
  const float* Wp  = (const float*)d_in[8];
  const float* bp  = (const float*)d_in[9];
  float* out = (float*)d_out;

  char* ws = (char*)d_ws;
  const size_t MB = 1u << 20;
  _Float16* Ah  = (_Float16*)(ws);            // x f16           8 MB
  _Float16* WqT = (_Float16*)(ws +  8 * MB);  // 2 MB each
  _Float16* WkT = (_Float16*)(ws + 10 * MB);
  _Float16* WvT = (_Float16*)(ws + 12 * MB);
  _Float16* WpT = (_Float16*)(ws + 14 * MB);
  _Float16* Qh  = (_Float16*)(ws + 16 * MB);  // 8 MB each
  _Float16* Kh  = (_Float16*)(ws + 24 * MB);
  _Float16* Vh  = (_Float16*)(ws + 32 * MB);
  _Float16* Fh  = (_Float16*)(ws + 40 * MB);  // feats, 8 MB -> 48 MB total

  f32_to_f16_kernel<<<(MROWS * HDIM) / 256, 256, 0, stream>>>(x, Ah, MROWS * HDIM);
  w_transpose_f16_kernel<<<(HDIM * HDIM) / 256, 256, 0, stream>>>(Wq, WqT);
  w_transpose_f16_kernel<<<(HDIM * HDIM) / 256, 256, 0, stream>>>(Wk, WkT);
  w_transpose_f16_kernel<<<(HDIM * HDIM) / 256, 256, 0, stream>>>(Wv, WvT);
  w_transpose_f16_kernel<<<(HDIM * HDIM) / 256, 256, 0, stream>>>(Wp, WpT);

  dim3 ggrid(HDIM / 128, MROWS / 128);  // (8, 32)
  gemm_f16_kernel<<<ggrid, 256, 0, stream>>>(Ah, WqT, bq, nullptr, Qh, 1, SCALE_F);
  gemm_f16_kernel<<<ggrid, 256, 0, stream>>>(Ah, WkT, bk, nullptr, Kh, 1, 1.0f);
  gemm_f16_kernel<<<ggrid, 256, 0, stream>>>(Ah, WvT, bv, nullptr, Vh, 1, 1.0f);

  attn_kernel<<<dim3(BB * NHEAD, NSEQ / 64), 128, 0, stream>>>(Qh, Kh, Vh, ab, Fh);

  gemm_f16_kernel<<<ggrid, 256, 0, stream>>>(Fh, WpT, bp, out, nullptr, 0, 1.0f);
}
